// RGATEncoder_69389491634570
// MI455X (gfx1250) — matrix operand order repaired and checked by reading, not compiled, not run on hardware
//
#include <hip/hip_runtime.h>

// Problem constants (from reference)
#define NN   100000   // nodes
#define RR   3        // relations
#define EE   500000   // edges per relation
#define HH   4        // heads
#define FF   64       // feat per head
#define INF_ 256      // in_feat
#define HF   256      // H*F
#define NEG_SLOPE 0.2f

typedef __attribute__((ext_vector_type(16))) __bf16       v16bf;
typedef __attribute__((ext_vector_type(8)))  float        v8f;
typedef __attribute__((ext_vector_type(4)))  unsigned int v4u;

union Frag {                 // one WMMA 16x32 bf16 operand fragment (per lane)
    v4u          q[2];
    unsigned int u[8];
    v16bf        v;
};

__device__ __forceinline__ unsigned short f32_to_bf16(float f) {
    unsigned int u = __float_as_uint(f);
    unsigned int r = u + 0x7FFFu + ((u >> 16) & 1u);   // round-to-nearest-even
    return (unsigned short)(r >> 16);
}

// Order-preserving float -> uint mapping (for atomicMax on floats incl. negatives)
__device__ __forceinline__ unsigned int ord_encode(float f) {
    unsigned int u = __float_as_uint(f);
    return (u & 0x80000000u) ? ~u : (u | 0x80000000u);
}
__device__ __forceinline__ float ord_decode(unsigned int u) {
    unsigned int v = (u & 0x80000000u) ? (u & 0x7FFFFFFFu) : ~u;
    return __uint_as_float(v);
}

// ---------------- casts / init ----------------

__global__ void cast_f32_to_bf16(const float* __restrict__ in,
                                 unsigned short* __restrict__ out, int n) {
    int t = blockIdx.x * blockDim.x + threadIdx.x;
    if (t < n) out[t] = f32_to_bf16(in[t]);
}

__global__ void zero_f32(float* __restrict__ p, int n) {
    int t = blockIdx.x * blockDim.x + threadIdx.x;
    if (t < n) p[t] = 0.f;
}

// ---------------- W pre-swizzle into WMMA B-fragment layout ----------------
// Packed layout per relation: frag = kblk*16 + tileN (kblk: K/32, tileN: col/16);
// within a frag: 32 lanes x 16 bf16 contiguous (32B per lane -> 2x b128 loads).
// Lane mapping (this code's contiguous k order for B, same as A): lane half 0 -> K 0..15,
// half 1 -> K 16..31; VGPR j holds K = half*16 + {2j, 2j+1}; col = tileN*16 + lane&15.
__global__ void pack_W_bfrag(const float* __restrict__ W,
                             unsigned short* __restrict__ wp) {
    int t = blockIdx.x * blockDim.x + threadIdx.x;
    if (t >= RR * INF_ * HF) return;
    int r     = t / (INF_ * HF);
    int i     = t - r * (INF_ * HF);
    int j     = i & 15;           // bf16 index within lane
    int lane  = (i >> 4) & 31;
    int tileN = (i >> 9) & 15;
    int kblk  = (i >> 13) & 7;
    int hf  = lane >> 4, l16 = lane & 15;
    int jj  = j >> 1,    sub = j & 1;
    int srcK   = kblk * 32 + hf * 16 + 2 * jj + sub;
    int srcCol = tileN * 16 + l16;
    wp[t] = f32_to_bf16(W[(size_t)r * INF_ * HF + (size_t)srcK * HF + srcCol]);
}

// ---------------- projection GEMM via WMMA bf16 ----------------
// z[100000,256] = x_bf16[100000,256] @ Wr_bf16[256,256]
// one wave computes a 16x16 tile; K consumed 32 at a time (8 WMMA ops).
// A: 2x b128 from row-major x_bf16; B: 2x b128 from pre-swizzled fragments.
__global__ void gemm_bf16_wmma(const unsigned short* __restrict__ xb,
                               const unsigned short* __restrict__ wp, // packed W_r
                               float* __restrict__ z) {
    int wave  = (int)((blockIdx.x * blockDim.x + threadIdx.x) >> 5);
    int lane  = threadIdx.x & 31;
    int tileN = wave & 15;        // 16 N-tiles (HF=256)
    int tileM = wave >> 4;        // 6250 M-tiles (N=100000)
    int hf    = lane >> 4;        // lane half
    int l16   = lane & 15;
    int row     = tileM * 16 + l16;
    int colBase = tileN * 16;

    v8f c = {};
    const unsigned short* arow = xb + (size_t)row * INF_;

#pragma unroll
    for (int kblk = 0; kblk < 8; ++kblk) {
        Frag a, b;
        // A (16x32 bf16): half 0 -> K 0..15 ; half 1 -> K 16..31
        a.q[0] = *(const v4u*)(arow + kblk * 32 + hf * 16);
        a.q[1] = *(const v4u*)(arow + kblk * 32 + hf * 16 + 8);
        // B: contiguous pre-swizzled fragment, 32B per lane
        const v4u* bp = (const v4u*)(wp + ((size_t)(kblk * 16 + tileN) * 32 + lane) * 16);
        b.q[0] = bp[0];
        b.q[1] = bp[1];
        c = __builtin_amdgcn_wmma_f32_16x16x32_bf16(
                false, a.v, false, b.v, (short)0, c, false, false);
    }

#pragma unroll
    for (int v = 0; v < 8; ++v) {
        int r = tileM * 16 + hf * 8 + v;     // C layout: VGPR v -> M = v + 8*half
        z[(size_t)r * HF + colBase + l16] = c[v];
    }
}

// ---------------- per-node attention logits ----------------
// el[n,h] = dot(z[n,h,:], al[h,:]) ; er likewise. Also zero-init m/den.
__global__ void attn_logits(const float* __restrict__ z,
                            const float* __restrict__ al,
                            const float* __restrict__ ar,
                            float* __restrict__ el, float* __restrict__ er,
                            unsigned int* __restrict__ mord,
                            float* __restrict__ den) {
    int t = blockIdx.x * blockDim.x + threadIdx.x;
    if (t >= NN * HH) return;
    int n = t >> 2, h = t & 3;
    const float* zr  = z  + (size_t)n * HF + h * FF;
    const float* alr = al + h * FF;
    const float* arr = ar + h * FF;
    float sl = 0.f, sr = 0.f;
#pragma unroll 8
    for (int f = 0; f < FF; ++f) {
        float zv = zr[f];
        sl += zv * alr[f];
        sr += zv * arr[f];
    }
    el[t] = sl; er[t] = sr;
    mord[t] = 0u;          // ord-encoded floor (< encode of any finite float)
    den[t]  = 0.f;
}

__device__ __forceinline__ float edge_score(const float* el, const float* er,
                                            int s, int d, int h) {
    float sc = el[s * HH + h] + er[d * HH + h];
    return sc > 0.f ? sc : NEG_SLOPE * sc;
}

// ---------------- edge pass 1: segment max ----------------
__global__ void edge_max(const int* __restrict__ src, const int* __restrict__ dst,
                         const float* __restrict__ el, const float* __restrict__ er,
                         unsigned int* __restrict__ mord) {
    int t = blockIdx.x * blockDim.x + threadIdx.x;
    if (t >= EE * HH) return;
    int e = t >> 2, h = t & 3;
    int s = src[e], d = dst[e];
    float sc = edge_score(el, er, s, d, h);
    atomicMax(&mord[d * HH + h], ord_encode(sc));
}

// ---------------- edge pass 2: exp + segment sum ----------------
__global__ void edge_exp(const int* __restrict__ src, const int* __restrict__ dst,
                         const float* __restrict__ el, const float* __restrict__ er,
                         const unsigned int* __restrict__ mord,
                         float* __restrict__ den, float* __restrict__ alpha) {
    int t = blockIdx.x * blockDim.x + threadIdx.x;
    if (t >= EE * HH) return;
    int e = t >> 2, h = t & 3;
    int s = src[e], d = dst[e];
    float sc = edge_score(el, er, s, d, h);
    float m  = ord_decode(mord[d * HH + h]);
    float ee = __expf(sc - m);
    alpha[t] = ee;
    atomicAdd(&den[d * HH + h], ee);
}

// ---------------- edge pass 3: normalize ----------------
__global__ void edge_norm(const int* __restrict__ dst,
                          const float* __restrict__ den,
                          float* __restrict__ alpha) {
    int t = blockIdx.x * blockDim.x + threadIdx.x;
    if (t >= EE * HH) return;
    int e = t >> 2, h = t & 3;
    int d = dst[e];
    alpha[t] = alpha[t] / den[d * HH + h];
}

// ---------------- edge pass 4: weighted scatter aggregate ----------------
// one 256-thread block per edge: out[dst, t] += alpha[h] * z[src, t]
__global__ void edge_aggregate(const int* __restrict__ src, const int* __restrict__ dst,
                               const float* __restrict__ z,
                               const float* __restrict__ alpha,
                               float* __restrict__ out) {
    int e = blockIdx.x;
    int t = threadIdx.x;
    int s = src[e], d = dst[e];           // uniform -> scalar loads
    int h = t >> 6;
    float a = alpha[e * HH + h];
    atomicAdd(&out[(size_t)d * HF + t], a * z[(size_t)s * HF + t]);
}

// ---------------- final: +bias sum, ELU in place ----------------
__global__ void bias_elu(const float* __restrict__ bias, float* __restrict__ out) {
    int t = blockIdx.x * blockDim.x + threadIdx.x;
    if (t >= NN * HF) return;
    int c = t & (HF - 1);
    float b = bias[c] + bias[HF + c] + bias[2 * HF + c];
    float v = out[t] + b;
    out[t] = v > 0.f ? v : (__expf(v) - 1.f);
}

extern "C" void kernel_launch(void* const* d_in, const int* in_sizes, int n_in,
                              void* d_out, int out_size, void* d_ws, size_t ws_size,
                              hipStream_t stream) {
    const float* x      = (const float*)d_in[0];   // [N, IN]
    const float* W      = (const float*)d_in[1];   // [R, IN, HF]
    const float* attn_l = (const float*)d_in[2];   // [R, H, F]
    const float* attn_r = (const float*)d_in[3];   // [R, H, F]
    const float* bias   = (const float*)d_in[4];   // [R, HF]
    const int*   src    = (const int*)d_in[5];     // [R, E]
    const int*   dst    = (const int*)d_in[6];     // [R, E]
    float* out = (float*)d_out;                    // [N, HF]
    (void)in_sizes; (void)n_in; (void)out_size; (void)ws_size;

    // Workspace carve-up (all sizes multiple of 256B)
    char* p = (char*)d_ws;
    unsigned short* xb   = (unsigned short*)p; p += (size_t)NN * INF_ * 2;        // 51.2 MB
    unsigned short* Wb   = (unsigned short*)p; p += (size_t)RR * INF_ * HF * 2;   // 384 KB (B-frag layout)
    float*          z    = (float*)p;          p += (size_t)NN * HF * 4;          // 102.4 MB
    float*          el   = (float*)p;          p += (size_t)NN * HH * 4;
    float*          er   = (float*)p;          p += (size_t)NN * HH * 4;
    unsigned int*   mord = (unsigned int*)p;   p += (size_t)NN * HH * 4;
    float*          den  = (float*)p;          p += (size_t)NN * HH * 4;
    float*          alpha= (float*)p;          p += (size_t)EE * HH * 4;          // 8 MB

    const int B = 256;

    // One-time: cast x, pack W into WMMA B-fragment layout, zero accumulator
    cast_f32_to_bf16<<<(NN * INF_ + B - 1) / B, B, 0, stream>>>(x, xb, NN * INF_);
    pack_W_bfrag<<<(RR * INF_ * HF + B - 1) / B, B, 0, stream>>>(W, Wb);
    zero_f32<<<(NN * HF + B - 1) / B, B, 0, stream>>>(out, NN * HF);

    for (int r = 0; r < RR; ++r) {
        const unsigned short* Wr = Wb + (size_t)r * INF_ * HF;
        const float* alr = attn_l + (size_t)r * HH * FF;
        const float* arr = attn_r + (size_t)r * HH * FF;
        const int* sr = src + (size_t)r * EE;
        const int* dr = dst + (size_t)r * EE;

        // z = x @ W_r : 100000 waves = 12500 blocks of 8 wave32 waves
        gemm_bf16_wmma<<<(NN / 16) * (HF / 16) / 8, B, 0, stream>>>(xb, Wr, z);

        attn_logits<<<(NN * HH + B - 1) / B, B, 0, stream>>>(z, alr, arr, el, er, mord, den);

        int eg = (EE * HH + B - 1) / B;
        edge_max <<<eg, B, 0, stream>>>(sr, dr, el, er, mord);
        edge_exp <<<eg, B, 0, stream>>>(sr, dr, el, er, mord, den, alpha);
        edge_norm<<<eg, B, 0, stream>>>(dr, den, alpha);

        edge_aggregate<<<EE, B, 0, stream>>>(sr, dr, z, alpha, out);
    }

    bias_elu<<<(NN * HF + B - 1) / B, B, 0, stream>>>(bias, out);
}
